// VectorQuantizer_2388001817302
// MI455X (gfx1250) — matrix operand order repaired and checked by reading, not compiled, not run on hardware
//
#include <hip/hip_runtime.h>

typedef __attribute__((ext_vector_type(16))) _Float16 v16h;
typedef __attribute__((ext_vector_type(8)))  _Float16 v8h;
typedef __attribute__((ext_vector_type(4)))  _Float16 v4h;
typedef __attribute__((ext_vector_type(8)))  float    v8f;

#define D_DIM          64
#define K_CODES        1024
#define LDS_STRIDE_H   72      // padded f16 row stride (144B): 36-dword stride -> conflict-free b128
#define TOK_PER_B      4096    // 64*64 spatial positions per batch image
#define N_TOK          65536   // 16 * 4096
#define ROWS_PER_BLOCK 128     // 8 waves * 16 rows

// ---------------------------------------------------------------------------
// Score one 16x16 tile: S = X(16x64) * E^T(64x16) in split-f16:
//   S ~= aH*bH + aH*bL + aL*bH   (f32 accumulate, ~2^-21 relative error)
// ---------------------------------------------------------------------------
__device__ __forceinline__ v8f tile_score(const v16h* aH, const v16h* aL,
                                          const _Float16* bh_base,
                                          const _Float16* bl_base)
{
    v8f c = {};
#pragma unroll
    for (int chunk = 0; chunk < 2; ++chunk) {
        const _Float16* ph = bh_base + chunk * 32;
        const _Float16* pl = bl_base + chunk * 32;
        v8h h0 = *(const v8h*)(ph);
        v8h h1 = *(const v8h*)(ph + 8);
        v8h l0 = *(const v8h*)(pl);
        v8h l1 = *(const v8h*)(pl + 8);
        v16h bh = __builtin_shufflevector(h0, h1, 0,1,2,3,4,5,6,7,8,9,10,11,12,13,14,15);
        v16h bl = __builtin_shufflevector(l0, l1, 0,1,2,3,4,5,6,7,8,9,10,11,12,13,14,15);
        c = __builtin_amdgcn_wmma_f32_16x16x32_f16(false, aH[chunk], false, bh,
                                                   (short)0, c, false, false);
        c = __builtin_amdgcn_wmma_f32_16x16x32_f16(false, aH[chunk], false, bl,
                                                   (short)0, c, false, false);
        c = __builtin_amdgcn_wmma_f32_16x16x32_f16(false, aL[chunk], false, bh,
                                                   (short)0, c, false, false);
    }
    return c;
}

// ---------------------------------------------------------------------------
// Kernel 1: distances via split-f16 WMMA + fused argmin over all 1024 codes.
// ---------------------------------------------------------------------------
__global__ void vq_argmin_kernel(const float* __restrict__ x,
                                 const float* __restrict__ cb,
                                 int*   __restrict__ idx_ws,
                                 float* __restrict__ idx_out)
{
    extern __shared__ char smem_raw[];
    _Float16* s_hi  = (_Float16*)smem_raw;                    // [1024][72] f16 hi plane
    _Float16* s_lo  = s_hi + K_CODES * LDS_STRIDE_H;          // [1024][72] f16 lo plane
    float*    s_esq = (float*)(s_lo + K_CODES * LDS_STRIDE_H);// [1024] f32 ||e||^2

    const int tid = threadIdx.x;

    // ---- stage codebook (split f16) + e_sq into LDS, once per workgroup ----
    for (int r = tid; r < K_CODES; r += blockDim.x) {
        const float* src = cb + r * D_DIM;
        _Float16* dh = s_hi + r * LDS_STRIDE_H;
        _Float16* dl = s_lo + r * LDS_STRIDE_H;
        float sq = 0.f;
#pragma unroll
        for (int d = 0; d < D_DIM; d += 4) {
            float4 v = *(const float4*)(src + d);
            sq += v.x * v.x + v.y * v.y + v.z * v.z + v.w * v.w;
            v4h hh, ll;
            _Float16 h;
            h = (_Float16)v.x; hh[0] = h; ll[0] = (_Float16)(v.x - (float)h);
            h = (_Float16)v.y; hh[1] = h; ll[1] = (_Float16)(v.y - (float)h);
            h = (_Float16)v.z; hh[2] = h; ll[2] = (_Float16)(v.z - (float)h);
            h = (_Float16)v.w; hh[3] = h; ll[3] = (_Float16)(v.w - (float)h);
            *(v4h*)(dh + d) = hh;
            *(v4h*)(dl + d) = ll;
        }
        s_esq[r] = sq;
    }
    __syncthreads();

    const int lane = tid & 31;
    const int wave = tid >> 5;
    const int half = lane >> 4;
    const int l15  = lane & 15;
    const int rowbase = blockIdx.x * ROWS_PER_BLOCK + wave * 16;

    // ---- A fragments (16x32 f16, contiguous k per lane half), built from channels-first x ----
    // x[token=(b,s)][d] = x[b*D*4096 + d*4096 + s]
    const int row = rowbase + l15;
    const int b   = row >> 12;
    const int sp  = row & (TOK_PER_B - 1);
    const float* xp = x + (size_t)b * (D_DIM * TOK_PER_B) + sp;

    v16h aH[2] = {}, aL[2] = {};
#pragma unroll
    for (int chunk = 0; chunk < 2; ++chunk) {
#pragma unroll
        for (int i = 0; i < 16; ++i) {
            // v16h element i holds K = half*16 + i (contiguous, matches B)
            const int k = chunk * 32 + half * 16 + i;
            const float f = xp[(size_t)k * TOK_PER_B];
            const _Float16 h = (_Float16)f;
            aH[chunk][i] = h;
            aL[chunk][i] = (_Float16)(f - (float)h);
        }
    }

    float bestv[8];
    int   besti[8];
#pragma unroll
    for (int i = 0; i < 8; ++i) { bestv[i] = 3.0e38f; besti[i] = 0; }

    // ---- sweep 64 column tiles, two per iteration (two independent chains) ----
    for (int ct = 0; ct < K_CODES / 16; ct += 2) {
        const int col0 = ct * 16 + l15;
        const int col1 = col0 + 16;
        // B (32x16 f16): lane half h holds K = h*16 + {0..15} contiguous
        const _Float16* bh0 = s_hi + col0 * LDS_STRIDE_H + half * 16;
        const _Float16* bl0 = s_lo + col0 * LDS_STRIDE_H + half * 16;
        const _Float16* bh1 = s_hi + col1 * LDS_STRIDE_H + half * 16;
        const _Float16* bl1 = s_lo + col1 * LDS_STRIDE_H + half * 16;

        const v8f c0 = tile_score(aH, aL, bh0, bl0);
        const v8f c1 = tile_score(aH, aL, bh1, bl1);

        const float esq0 = s_esq[col0];
        const float esq1 = s_esq[col1];
#pragma unroll
        for (int i = 0; i < 8; ++i) {   // VGPR i holds rows i (lanes 0-15) / i+8 (16-31)
            const float v0 = esq0 - 2.0f * c0[i];
            if (v0 < bestv[i]) { bestv[i] = v0; besti[i] = col0; }
            const float v1 = esq1 - 2.0f * c1[i];
            if (v1 < bestv[i]) { bestv[i] = v1; besti[i] = col1; }
        }
    }

    // ---- cross-lane min within each 16-lane half (same 8 rows per half) ----
#pragma unroll
    for (int m = 1; m < 16; m <<= 1) {
#pragma unroll
        for (int i = 0; i < 8; ++i) {
            const float ov = __shfl_xor(bestv[i], m, 32);
            const int   oi = __shfl_xor(besti[i], m, 32);
            if (ov < bestv[i] || (ov == bestv[i] && oi < besti[i])) {
                bestv[i] = ov; besti[i] = oi;
            }
        }
    }

    if (l15 == 0) {
#pragma unroll
        for (int i = 0; i < 8; ++i) {
            const int r = rowbase + half * 8 + i;
            idx_ws[r]  = besti[i];
            idx_out[r] = (float)besti[i];
        }
    }
}

// ---------------------------------------------------------------------------
// Kernel 2: channels-first gather  out[b,d,h,w] = cb[idx[b,h,w], d]
// Coalesced writes; 256KB codebook is L2-resident (192MB L2).
// ---------------------------------------------------------------------------
__global__ void vq_gather_kernel(const float* __restrict__ cb,
                                 const int*   __restrict__ idx,
                                 float* __restrict__ out)
{
    const int o  = blockIdx.x * blockDim.x + threadIdx.x;
    const int sp = o & 4095;
    const int d  = (o >> 12) & 63;
    const int bb = o >> 18;
    const int token = (bb << 12) | sp;
    out[o] = cb[idx[token] * D_DIM + d];
}

extern "C" void kernel_launch(void* const* d_in, const int* in_sizes, int n_in,
                              void* d_out, int out_size, void* d_ws, size_t ws_size,
                              hipStream_t stream)
{
    const float* x  = (const float*)d_in[0];   // (16, 64, 64, 64) f32
    const float* cb = (const float*)d_in[1];   // (1024, 64) f32

    float* out       = (float*)d_out;                         // quantized output
    float* idx_out_f = out + (size_t)16 * 64 * 64 * 64;       // tail: indices as float
    int*   idx_ws    = (int*)d_ws;                            // int indices for gather

    const size_t lds_bytes = (size_t)2 * K_CODES * LDS_STRIDE_H * sizeof(_Float16)
                           + (size_t)K_CODES * sizeof(float);  // 292 KB

    vq_argmin_kernel<<<N_TOK / ROWS_PER_BLOCK, 256, lds_bytes, stream>>>(
        x, cb, idx_ws, idx_out_f);

    const int total = 16 * 64 * 64 * 64;
    vq_gather_kernel<<<total / 256, 256, 0, stream>>>(cb, idx_ws, out);
}
